// GCN2_84413287235625
// MI455X (gfx1250) — matrix operand order repaired and checked by reading, not compiled, not run on hardware
//
#include <hip/hip_runtime.h>
#include <hip/hip_bf16.h>
#include <math.h>

#define N_NODES 100000
#define N_EDGES 1600000
#define N_GRAPH 128
#define DIN     771
#define DH      256
#define NL      8
#define K_PAD   800   // 25 * 32, zero padded beyond 771

typedef __attribute__((ext_vector_type(8)))  float  v8f;
typedef __attribute__((ext_vector_type(16))) __bf16 v16bf;
typedef __attribute__((ext_vector_type(8)))  __bf16 v8bf;
typedef __attribute__((ext_vector_type(4)))  __bf16 v4bf;

__device__ __forceinline__ void atomAddF(float* p, float v) {
  __hip_atomic_fetch_add(p, v, __ATOMIC_RELAXED, __HIP_MEMORY_SCOPE_AGENT);
}

// ---------------- degree / norm ----------------
__global__ void fill_ones_kernel(float* p, int n) {
  int i = blockIdx.x * blockDim.x + threadIdx.x;
  if (i < n) p[i] = 1.0f;   // self-loop contributes 1 to every degree
}
__global__ void deg_accum_kernel(const int* __restrict__ dst, float* __restrict__ deg, int e) {
  int i = blockIdx.x * blockDim.x + threadIdx.x;
  if (i < e) atomAddF(&deg[dst[i]], 1.0f);
}
__global__ void dinv_kernel(float* p, int n) {
  int i = blockIdx.x * blockDim.x + threadIdx.x;
  if (i < n) { float d = p[i]; p[i] = (d > 0.f) ? rsqrtf(d) : 0.f; }
}

// ------------- weights -> transposed bf16 -------------
__global__ void convert_weights_kernel(const float* __restrict__ winf,
                                       const float* __restrict__ w1f,
                                       const float* __restrict__ w2f,
                                       __bf16* __restrict__ WinT,
                                       __bf16* __restrict__ W1T,
                                       __bf16* __restrict__ W2T) {
  int idx = blockIdx.x * blockDim.x + threadIdx.x;
  const int winN = 256 * K_PAD;          // 204800
  const int wN   = NL * 256 * 256;       // 524288
  if (idx < winN) {
    int n = idx / K_PAD, k = idx % K_PAD;
    WinT[idx] = (k < DIN) ? (__bf16)winf[(size_t)k * 256 + n] : (__bf16)0.f;
  } else if (idx < winN + wN) {
    int j = idx - winN;
    int l = j >> 16, rem = j & 65535, n = rem >> 8, k = rem & 255;
    W1T[j] = (__bf16)w1f[(size_t)l * 65536 + (size_t)k * 256 + n];
  } else if (idx < winN + 2 * wN) {
    int j = idx - winN - wN;
    int l = j >> 16, rem = j & 65535, n = rem >> 8, k = rem & 255;
    W2T[j] = (__bf16)w2f[(size_t)l * 65536 + (size_t)k * 256 + n];
  }
}

// Build WMMA 16-bit A-fragment from a row-major bf16 LDS row.
// Lane layout (contiguous, matches B): low half-wave K[kb..kb+15],
// high half-wave K[kb+16..kb+31] -> two contiguous 16B chunks.
__device__ __forceinline__ v16bf load_afrag(const __bf16* row, int kb, int half) {
  const int k0 = kb + half * 16;
  const int k1 = kb + half * 16 + 8;
  v8bf lo = *(const v8bf*)(row + k0);
  v8bf hi = *(const v8bf*)(row + k1);
  return __builtin_shufflevector(lo, hi, 0, 1, 2, 3, 4, 5, 6, 7,
                                 8, 9, 10, 11, 12, 13, 14, 15);
}

// ------------- input GEMM: h = h0 = relu(x @ Win + b) -------------
// block = 8 waves; block tile = 16 rows x 256 cols; wave tile = 16 x 32.
// A tile (16 x 800, bf16) staged once per block in LDS.
__global__ __launch_bounds__(256) void gemm_in_kernel(
    const float* __restrict__ X, const __bf16* __restrict__ WinT,
    const float* __restrict__ bias, float* __restrict__ hA, float* __restrict__ hB)
{
  __shared__ __bf16 xLds[16 * K_PAD];   // 25.6 KB
  const int tid  = threadIdx.x;
  const int lane = tid & 31;
  const int wave = tid >> 5;
  const int half = lane >> 4;
  const int m    = lane & 15;
  const int row0 = blockIdx.x * 16;
  const int n0   = wave * 32 + m;

  // cooperative stage: coalesced global f32 -> bf16 LDS, zero pad K in [771,800)
  for (int idx = tid; idx < 16 * K_PAD; idx += 256) {
    int r = idx / K_PAD, k = idx - r * K_PAD;
    float v = (k < DIN) ? X[(size_t)(row0 + r) * DIN + k] : 0.f;
    xLds[idx] = (__bf16)v;
  }
  __syncthreads();

  v8f acc0 = {}; v8f acc1 = {};
  const __bf16* arow = xLds + m * K_PAD;
  const __bf16* wb0  = WinT + (size_t)n0 * K_PAD;
  const __bf16* wb1  = WinT + (size_t)(n0 + 16) * K_PAD;

  for (int kc = 0; kc < 25; ++kc) {
    const int kb = kc * 32;
    v16bf aF = load_afrag(arow, kb, half);
    const int ks = kb + half * 16;       // B: lane=col, low half K[kb..+15], high K[kb+16..+31]
    v16bf b0 = *(const v16bf*)(wb0 + ks);
    v16bf b1 = *(const v16bf*)(wb1 + ks);
    acc0 = __builtin_amdgcn_wmma_f32_16x16x32_bf16(false, aF, false, b0, (short)0, acc0, false, false);
    acc1 = __builtin_amdgcn_wmma_f32_16x16x32_bf16(false, aF, false, b1, (short)0, acc1, false, false);
  }
  const float bi0 = bias[n0], bi1 = bias[n0 + 16];
#pragma unroll
  for (int v = 0; v < 8; ++v) {
    int r = row0 + v + half * 8;         // C layout: M = vgpr + 8*half, N = lane&15
    size_t o0 = (size_t)r * DH + n0;
    float a = fmaxf(acc0[v] + bi0, 0.f);
    float c = fmaxf(acc1[v] + bi1, 0.f);
    hA[o0] = a;      hB[o0] = a;
    hA[o0 + 16] = c; hB[o0 + 16] = c;
  }
}

// ------------- SpMM: agg[dst] += norm * h[src] (one wave per edge) -------------
__global__ __launch_bounds__(256) void spmm_kernel(
    const float* __restrict__ h, const float* __restrict__ dinv,
    const int* __restrict__ src, const int* __restrict__ dst,
    float* __restrict__ agg)
{
  const int lane = threadIdx.x & 31;
  const int wave = threadIdx.x >> 5;
  const long long e = (long long)blockIdx.x * 8 + wave;
  int s, d; float w;
  if (e < N_EDGES) {
    s = src[e]; d = dst[e];
    w = dinv[s] * dinv[d];
  } else {                               // self loops appended after edges
    s = d = (int)(e - N_EDGES);
    float di = dinv[s];
    w = di * di;
  }
  const float4* hr = (const float4*)(h + (size_t)s * DH);
  float4 p0 = hr[lane];
  float4 p1 = hr[lane + 32];
  float* ag = agg + (size_t)d * DH;
  int c = 4 * lane;
  atomAddF(ag + c + 0, w * p0.x); atomAddF(ag + c + 1, w * p0.y);
  atomAddF(ag + c + 2, w * p0.z); atomAddF(ag + c + 3, w * p0.w);
  atomAddF(ag + 128 + c + 0, w * p1.x); atomAddF(ag + 128 + c + 1, w * p1.y);
  atomAddF(ag + 128 + c + 2, w * p1.z); atomAddF(ag + 128 + c + 3, w * p1.w);
}

// ------------- layer: out = (1-b)*(0.9agg+0.1h0) + b*(0.9agg@W1 + 0.1h0@W2); LN; relu -------------
__global__ __launch_bounds__(256) void layer_kernel(
    const float* __restrict__ agg, const float* __restrict__ h0,
    const __bf16* __restrict__ W1T, const __bf16* __restrict__ W2T,
    const float* __restrict__ lng, const float* __restrict__ lnb,
    float* __restrict__ hOut, float beta)
{
  __shared__ __bf16 aLds[16 * DH];      // 0.9*agg, 8 KB
  __shared__ __bf16 rLds[16 * DH];      // 0.1*h0,  8 KB
  __shared__ float  oLds[16 * 257];     // pre-LN output, 16.4 KB
  const int tid  = threadIdx.x;
  const int lane = tid & 31;
  const int wave = tid >> 5;
  const int half = lane >> 4;
  const int m    = lane & 15;
  const int row0 = blockIdx.x * 16;
  const int cb   = wave * 32;

  // cooperative stage: 1024 float4s per array, 4 per thread, packed v4bf stores
  for (int idx = tid; idx < 16 * DH / 4; idx += 256) {
    int r = idx >> 6, c4 = idx & 63;
    size_t go = (size_t)(row0 + r) * DH + c4 * 4;
    float4 a = *(const float4*)(agg + go);
    float4 h = *(const float4*)(h0 + go);
    int base = r * DH + c4 * 4;
    *(v4bf*)(aLds + base) = (v4bf){(__bf16)(0.9f * a.x), (__bf16)(0.9f * a.y),
                                   (__bf16)(0.9f * a.z), (__bf16)(0.9f * a.w)};
    *(v4bf*)(rLds + base) = (v4bf){(__bf16)(0.1f * h.x), (__bf16)(0.1f * h.y),
                                   (__bf16)(0.1f * h.z), (__bf16)(0.1f * h.w)};
  }
  __syncthreads();

  v8f acc[2] = {{}, {}};
  const __bf16* arow = aLds + m * DH;
  const __bf16* rrow = rLds + m * DH;
#pragma unroll
  for (int kc = 0; kc < 8; ++kc) {
    const int kb = kc * 32;
    v16bf aF = load_afrag(arow, kb, half);
    v16bf rF = load_afrag(rrow, kb, half);
    const int ks = kb + half * 16;
#pragma unroll
    for (int t = 0; t < 2; ++t) {
      int n = cb + t * 16 + m;
      v16bf b1 = *(const v16bf*)(W1T + (size_t)n * DH + ks);
      v16bf b2 = *(const v16bf*)(W2T + (size_t)n * DH + ks);
      acc[t] = __builtin_amdgcn_wmma_f32_16x16x32_bf16(false, aF, false, b1, (short)0, acc[t], false, false);
      acc[t] = __builtin_amdgcn_wmma_f32_16x16x32_bf16(false, rF, false, b2, (short)0, acc[t], false, false);
    }
  }
  const float omb = 1.0f - beta;
#pragma unroll
  for (int t = 0; t < 2; ++t) {
    int col = cb + t * 16 + m;
#pragma unroll
    for (int v = 0; v < 8; ++v) {
      int r = v + half * 8;
      float s = (float)aLds[r * DH + col] + (float)rLds[r * DH + col];
      oLds[r * 257 + col] = omb * s + beta * acc[t][v];
    }
  }
  __syncthreads();
  // LayerNorm + ReLU: each wave owns 2 rows, wave32 shuffle reduction
  for (int rr = 0; rr < 2; ++rr) {
    int r = wave * 2 + rr;
    float vals[8]; float sum = 0.f, sq = 0.f;
#pragma unroll
    for (int i = 0; i < 8; ++i) {
      float v = oLds[r * 257 + lane + 32 * i];
      vals[i] = v; sum += v; sq += v * v;
    }
#pragma unroll
    for (int off = 16; off > 0; off >>= 1) {
      sum += __shfl_xor(sum, off, 32);
      sq  += __shfl_xor(sq,  off, 32);
    }
    float mean = sum * (1.0f / 256.0f);
    float var  = sq * (1.0f / 256.0f) - mean * mean;
    float rstd = rsqrtf(var + 1e-5f);
#pragma unroll
    for (int i = 0; i < 8; ++i) {
      int c = lane + 32 * i;
      float oo = (vals[i] - mean) * rstd * lng[c] + lnb[c];
      hOut[(size_t)(row0 + r) * DH + c] = fmaxf(oo, 0.f);
    }
  }
}

// ------------- pooling + head -------------
__global__ __launch_bounds__(256) void pool_kernel(
    const float* __restrict__ h, const int* __restrict__ batch,
    float* __restrict__ pooled, float* __restrict__ cnt)
{
  const int lane = threadIdx.x & 31;
  const int wave = threadIdx.x >> 5;
  const int node = blockIdx.x * 8 + wave;
  if (node >= N_NODES) return;
  const int b = batch[node];
  const float4* hr = (const float4*)(h + (size_t)node * DH);
  float4 p0 = hr[lane], p1 = hr[lane + 32];
  float* pg = pooled + (size_t)b * DH;
  int c = 4 * lane;
  atomAddF(pg + c + 0, p0.x); atomAddF(pg + c + 1, p0.y);
  atomAddF(pg + c + 2, p0.z); atomAddF(pg + c + 3, p0.w);
  atomAddF(pg + 128 + c + 0, p1.x); atomAddF(pg + 128 + c + 1, p1.y);
  atomAddF(pg + 128 + c + 2, p1.z); atomAddF(pg + 128 + c + 3, p1.w);
  if (lane == 0) atomAddF(cnt + b, 1.0f);
}

__global__ __launch_bounds__(256) void mlp1_kernel(
    const float* __restrict__ pooled, const float* __restrict__ cnt,
    const float* __restrict__ w, const float* __restrict__ b, float* __restrict__ z1)
{
  __shared__ float pr[256];
  const int g = blockIdx.x, j = threadIdx.x;
  float invc = 1.0f / fmaxf(cnt[g], 1.0f);
  pr[j] = pooled[(size_t)g * 256 + j] * invc;
  __syncthreads();
  float s = b[j];
  for (int k = 0; k < 256; ++k) s += pr[k] * w[(size_t)k * 256 + j];
  z1[(size_t)g * 256 + j] = fmaxf(s, 0.f);
}

__global__ __launch_bounds__(128) void mlp2_kernel(
    const float* __restrict__ z1, const float* __restrict__ w,
    const float* __restrict__ b, float* __restrict__ z2)
{
  __shared__ float zr[256];
  const int g = blockIdx.x, j = threadIdx.x;
  zr[j]       = z1[(size_t)g * 256 + j];
  zr[j + 128] = z1[(size_t)g * 256 + j + 128];
  __syncthreads();
  float s = b[j];
  for (int k = 0; k < 256; ++k) s += zr[k] * w[(size_t)k * 128 + j];
  z2[(size_t)g * 128 + j] = fmaxf(s, 0.f);
}

__global__ __launch_bounds__(128) void mlp3_kernel(
    const float* __restrict__ z2, const float* __restrict__ w,
    const float* __restrict__ b, float* __restrict__ out)
{
  const int g = threadIdx.x;
  float s = b[0];
  for (int k = 0; k < 128; ++k) s += z2[(size_t)g * 128 + k] * w[k];
  out[g] = s;
}

extern "C" void kernel_launch(void* const* d_in, const int* in_sizes, int n_in,
                              void* d_out, int out_size, void* d_ws, size_t ws_size,
                              hipStream_t stream)
{
  const float* x     = (const float*)d_in[0];
  const int*   ei    = (const int*)d_in[1];
  const int*   batch = (const int*)d_in[2];
  const float* winf  = (const float*)d_in[3];
  const float* winb  = (const float*)d_in[4];
  const float* w1f   = (const float*)d_in[5];
  const float* w2f   = (const float*)d_in[6];
  const float* lng   = (const float*)d_in[7];
  const float* lnb   = (const float*)d_in[8];
  const float* c1w   = (const float*)d_in[9];
  const float* c1b   = (const float*)d_in[10];
  const float* c2w   = (const float*)d_in[11];
  const float* c2b   = (const float*)d_in[12];
  const float* c3w   = (const float*)d_in[13];
  const float* c3b   = (const float*)d_in[14];
  float* out = (float*)d_out;

  char* ws = (char*)d_ws;
  size_t o = 0;
  auto take = [&](size_t bytes) -> char* {
    char* p = ws + o;
    o = (o + bytes + 511) & ~(size_t)511;
    return p;
  };
  float* dinv = (float*)take((size_t)N_NODES * 4);
  float* hA   = (float*)take((size_t)N_NODES * DH * 4);   // current h
  float* hB   = (float*)take((size_t)N_NODES * DH * 4);   // h0
  float* hC   = (float*)take((size_t)N_NODES * DH * 4);   // agg
  __bf16* WinT = (__bf16*)take((size_t)256 * K_PAD * 2);
  __bf16* W1T  = (__bf16*)take((size_t)NL * 65536 * 2);
  __bf16* W2T  = (__bf16*)take((size_t)NL * 65536 * 2);
  float* pooled = (float*)take((size_t)N_GRAPH * DH * 4);
  float* cnt    = (float*)take((size_t)N_GRAPH * 4);
  float* z1     = (float*)take((size_t)N_GRAPH * 256 * 4);
  float* z2     = (float*)take((size_t)N_GRAPH * 128 * 4);
  if (o > ws_size) return;

  const int* esrc = ei;
  const int* edst = ei + N_EDGES;

  // gcn_norm
  fill_ones_kernel<<<(N_NODES + 255) / 256, 256, 0, stream>>>(dinv, N_NODES);
  deg_accum_kernel<<<(N_EDGES + 255) / 256, 256, 0, stream>>>(edst, dinv, N_EDGES);
  dinv_kernel<<<(N_NODES + 255) / 256, 256, 0, stream>>>(dinv, N_NODES);

  // weights -> transposed bf16
  {
    int total = 256 * K_PAD + 2 * NL * 65536;
    convert_weights_kernel<<<(total + 255) / 256, 256, 0, stream>>>(winf, w1f, w2f, WinT, W1T, W2T);
  }

  // input layer
  gemm_in_kernel<<<N_NODES / 16, 256, 0, stream>>>(x, WinT, winb, hA, hB);

  // 8 GCN2 layers
  for (int li = 0; li < NL; ++li) {
    float beta = logf(0.5f / (float)(li + 1) + 1.0f);
    hipMemsetAsync(hC, 0, (size_t)N_NODES * DH * 4, stream);
    spmm_kernel<<<(N_EDGES + N_NODES) / 8, 256, 0, stream>>>(hA, dinv, esrc, edst, hC);
    layer_kernel<<<N_NODES / 16, 256, 0, stream>>>(
        hC, hB, W1T + (size_t)li * 65536, W2T + (size_t)li * 65536,
        lng + (size_t)li * 256, lnb + (size_t)li * 256, hA, beta);
  }

  // pool + head
  hipMemsetAsync(pooled, 0, (size_t)N_GRAPH * DH * 4, stream);
  hipMemsetAsync(cnt, 0, (size_t)N_GRAPH * 4, stream);
  pool_kernel<<<(N_NODES + 7) / 8, 256, 0, stream>>>(hA, batch, pooled, cnt);
  mlp1_kernel<<<N_GRAPH, 256, 0, stream>>>(pooled, cnt, c1w, c1b, z1);
  mlp2_kernel<<<N_GRAPH, 128, 0, stream>>>(z1, c2w, c2b, z2);
  mlp3_kernel<<<1, 128, 0, stream>>>(z2, c3w, c3b, out);
}
